// EEGGraphConvNetTemporal_27049704030290
// MI455X (gfx1250) — compile-verified, matrix operand order checked by reading, not run on hardware
//
#include <hip/hip_runtime.h>

typedef __attribute__((ext_vector_type(16))) __bf16 v16bf;
typedef __attribute__((ext_vector_type(8)))  __bf16 v8bf;
typedef __attribute__((ext_vector_type(8)))  float  v8f;

static constexpr int   NN   = 20000;
static constexpr int   EE   = 320000;
static constexpr int   F0   = 1280;
static constexpr int   NG   = 256;
static constexpr float EPSV = 1e-5f;

__device__ __forceinline__ float lrelu(float x) { return x >= 0.f ? x : 0.01f * x; }

__device__ __forceinline__ void atomAddF(float* p, float v) {
    __hip_atomic_fetch_add(p, v, __ATOMIC_RELAXED, __HIP_MEMORY_SCOPE_AGENT);
}

// ---------------- utility kernels ----------------

__global__ void k_zero_f32(float* __restrict__ p, long long n) {
    long long i = (long long)blockIdx.x * blockDim.x + threadIdx.x;
    if (i < n) p[i] = 0.f;
}

__global__ void k_f32_to_bf16(const float* __restrict__ in, __bf16* __restrict__ out, long long n) {
    long long i = (long long)blockIdx.x * blockDim.x + threadIdx.x;
    if (i < n) out[i] = (__bf16)in[i];
}

// Wt[fo*K + k] = bf16(W[k*Fo + fo])   (row-major K x Fo  ->  row-major Fo x K)
__global__ void k_transpose_bf16(const float* __restrict__ W, __bf16* __restrict__ Wt, int K, int Fo) {
    long long i = (long long)blockIdx.x * blockDim.x + threadIdx.x;
    long long total = (long long)K * Fo;
    if (i >= total) return;
    int fo = (int)(i / K);
    int k  = (int)(i % K);
    Wt[(long long)fo * K + k] = (__bf16)W[(long long)k * Fo + fo];
}

// ---------------- WMMA GEMM: C[N x Fo] = A[N x K](bf16) * Bt[Fo x K]^T(bf16) ----------------
// Register-blocked: each wave computes a 32x64 macro-tile (2 M-tiles x 4 N-tiles,
// 8 f32 accumulators). Per 32-wide k-step: 4 A-fragment + 8 B-fragment b128 loads
// feed 8 v_wmma_f32_16x16x32_bf16 (1.5 loads/WMMA; A reused 4x, B reused 2x in regs).
// Block = 2 waves -> block tile 32 x 128.  Grid: (NN/32, Fo/128); all dims divide exactly.

__global__ __launch_bounds__(64)
void k_gemm_bf16(const __bf16* __restrict__ A, const __bf16* __restrict__ Bt,
                 float* __restrict__ C, int K, int Fo) {
    const int wave  = threadIdx.x >> 5;
    const int lane  = threadIdx.x & 31;
    const int l15   = lane & 15;
    const int half  = (lane >> 4) & 1;
    const int tileM = blockIdx.x * 32;
    const int tileN = (blockIdx.y * 2 + wave) * 64;

    // A fragment rows (16x32 bf16 tile): lanes<16 hold K{0..7,16..23}; lanes>=16 hold K{8..15,24..31}
    const __bf16* arow0 = A  + (long long)(tileM + l15) * K;
    const __bf16* arow1 = A  + (long long)(tileM + 16 + l15) * K;
    const __bf16* brow0 = Bt + (long long)(tileN + l15) * K;
    const __bf16* brow1 = Bt + (long long)(tileN + 16 + l15) * K;
    const __bf16* brow2 = Bt + (long long)(tileN + 32 + l15) * K;
    const __bf16* brow3 = Bt + (long long)(tileN + 48 + l15) * K;

    v8f acc[2][4];
#pragma unroll
    for (int m = 0; m < 2; ++m)
#pragma unroll
        for (int n = 0; n < 4; ++n)
            acc[m][n] = (v8f){};

    for (int kk = 0; kk < K; kk += 32) {
        const int k0 = kk + half * 8;

        v8bf a0lo = *(const v8bf*)(arow0 + k0);
        v8bf a0hi = *(const v8bf*)(arow0 + k0 + 16);
        v8bf a1lo = *(const v8bf*)(arow1 + k0);
        v8bf a1hi = *(const v8bf*)(arow1 + k0 + 16);
        v8bf b0lo = *(const v8bf*)(brow0 + k0);
        v8bf b0hi = *(const v8bf*)(brow0 + k0 + 16);
        v8bf b1lo = *(const v8bf*)(brow1 + k0);
        v8bf b1hi = *(const v8bf*)(brow1 + k0 + 16);
        v8bf b2lo = *(const v8bf*)(brow2 + k0);
        v8bf b2hi = *(const v8bf*)(brow2 + k0 + 16);
        v8bf b3lo = *(const v8bf*)(brow3 + k0);
        v8bf b3hi = *(const v8bf*)(brow3 + k0 + 16);

        v16bf av[2], bv[4];
        av[0] = __builtin_shufflevector(a0lo, a0hi, 0,1,2,3,4,5,6,7,8,9,10,11,12,13,14,15);
        av[1] = __builtin_shufflevector(a1lo, a1hi, 0,1,2,3,4,5,6,7,8,9,10,11,12,13,14,15);
        bv[0] = __builtin_shufflevector(b0lo, b0hi, 0,1,2,3,4,5,6,7,8,9,10,11,12,13,14,15);
        bv[1] = __builtin_shufflevector(b1lo, b1hi, 0,1,2,3,4,5,6,7,8,9,10,11,12,13,14,15);
        bv[2] = __builtin_shufflevector(b2lo, b2hi, 0,1,2,3,4,5,6,7,8,9,10,11,12,13,14,15);
        bv[3] = __builtin_shufflevector(b3lo, b3hi, 0,1,2,3,4,5,6,7,8,9,10,11,12,13,14,15);

#pragma unroll
        for (int m = 0; m < 2; ++m)
#pragma unroll
            for (int n = 0; n < 4; ++n)
                acc[m][n] = __builtin_amdgcn_wmma_f32_16x16x32_bf16(
                    false, av[m], false, bv[n], (short)0, acc[m][n], false, false);
    }

    // C/D layout: VGPR r -> M = r (lanes 0-15) or 8+r (lanes 16-31); N = lane&15
#pragma unroll
    for (int m = 0; m < 2; ++m) {
#pragma unroll
        for (int n = 0; n < 4; ++n) {
            float* crow = C + (long long)(tileM + m * 16 + half * 8) * Fo
                            + tileN + n * 16 + l15;
#pragma unroll
            for (int r = 0; r < 8; ++r)
                crow[(long long)r * Fo] = acc[m][n][r];
        }
    }
}

// ---------------- edge scatter: agg[dst] += ew * H[src] ----------------

__global__ void k_scatter(const float* __restrict__ H, const int* __restrict__ src,
                          const int* __restrict__ dst, const float* __restrict__ ew,
                          float* __restrict__ agg, int E, int F) {
    long long t = (long long)blockIdx.x * blockDim.x + threadIdx.x;
    const int fq = F >> 2;
    long long total = (long long)E * fq;
    if (t >= total) return;
    int e = (int)(t / fq);
    int c = (int)(t % fq) * 4;
    float w = ew[e];
    int s = src[e], d = dst[e];
    const float4 hv = *(const float4*)(H + (long long)s * F + c);
    float* o = agg + (long long)d * F + c;
    atomAddF(o + 0, w * hv.x);
    atomAddF(o + 1, w * hv.y);
    atomAddF(o + 2, w * hv.z);
    atomAddF(o + 3, w * hv.w);
}

// ---------------- per-column sum / sum-of-squares ----------------

__global__ void k_colstats(const float* __restrict__ X, float* __restrict__ sums,
                           float* __restrict__ sumsq, int Nn, int F, int rowsPer) {
    int col = blockIdx.x * blockDim.x + threadIdx.x;
    if (col >= F) return;
    int r0 = blockIdx.y * rowsPer;
    int r1 = r0 + rowsPer; if (r1 > Nn) r1 = Nn;
    float s = 0.f, s2 = 0.f;
    for (int r = r0; r < r1; ++r) {
        float v = X[(long long)r * F + col];
        s += v; s2 += v * v;
    }
    atomAddF(&sums[col], s);
    atomAddF(&sumsq[col], s2);
}

// ---------------- fused BatchNorm (training stats) + LeakyReLU ----------------
// GCN bias b cancels inside BN (mean subtraction), so it is omitted.

__global__ void k_bn_act(const float* __restrict__ agg, const float* __restrict__ sums,
                         const float* __restrict__ sumsq, const float* __restrict__ g,
                         const float* __restrict__ be, float* __restrict__ actf,
                         __bf16* __restrict__ actb, int Nn, int F) {
    long long i = (long long)blockIdx.x * blockDim.x + threadIdx.x;
    long long total = (long long)Nn * F;
    if (i >= total) return;
    int col = (int)(i % F);
    float inv = 1.0f / (float)Nn;
    float m   = sums[col] * inv;
    float var = sumsq[col] * inv - m * m;
    float y = (agg[i] - m) * rsqrtf(var + EPSV) * g[col] + be[col];
    y = lrelu(y);
    actf[i] = y;
    actb[i] = (__bf16)y;
}

// ---------------- graph pooling: pooled[batch[n]] += h[n] ----------------

__global__ void k_pool(const float* __restrict__ H, const int* __restrict__ batch,
                       float* __restrict__ pooled, int Nn, int F) {
    long long i = (long long)blockIdx.x * blockDim.x + threadIdx.x;
    const int fq = F >> 2;
    long long total = (long long)Nn * fq;
    if (i >= total) return;
    int n = (int)(i / fq);
    int c = (int)(i % fq) * 4;
    int gid = batch[n];
    const float4 v = *(const float4*)(H + (long long)n * F + c);
    float* o = pooled + (long long)gid * F + c;
    atomAddF(o + 0, v.x);
    atomAddF(o + 1, v.y);
    atomAddF(o + 2, v.z);
    atomAddF(o + 3, v.w);
}

// ---------------- MLP head: one block per graph ----------------

__global__ __launch_bounds__(128)
void k_fc_head(const float* __restrict__ pooled,
               const float* __restrict__ Wf1, const float* __restrict__ bf1,
               const float* __restrict__ Wf2, const float* __restrict__ bf2,
               const float* __restrict__ Wf3, const float* __restrict__ bf3,
               float* __restrict__ out) {
    __shared__ float s_in[256];
    __shared__ float s_h1[128];
    __shared__ float s_h2[64];
    const int g = blockIdx.x;
    const int t = threadIdx.x;

    s_in[t]       = pooled[g * 256 + t];
    s_in[t + 128] = pooled[g * 256 + t + 128];
    __syncthreads();

    float a = bf1[t];
    for (int k = 0; k < 256; ++k) a += s_in[k] * Wf1[k * 128 + t];
    s_h1[t] = lrelu(a);
    __syncthreads();

    if (t < 64) {
        float a2 = bf2[t];
        for (int k = 0; k < 128; ++k) a2 += s_h1[k] * Wf2[k * 64 + t];
        s_h2[t] = lrelu(a2);
    }
    __syncthreads();

    if (t < 2) {
        float a3 = bf3[t];
        for (int k = 0; k < 64; ++k) a3 += s_h2[k] * Wf3[k * 2 + t];
        out[g * 2 + t] = lrelu(a3);
    }
}

// ---------------- host side ----------------

static inline size_t alignUp(size_t x) { return (x + 255) & ~size_t(255); }
static inline unsigned gblk(long long n, int b) { return (unsigned)((n + b - 1) / b); }

extern "C" void kernel_launch(void* const* d_in, const int* in_sizes, int n_in,
                              void* d_out, int out_size, void* d_ws, size_t ws_size,
                              hipStream_t stream) {
    (void)in_sizes; (void)n_in; (void)out_size; (void)ws_size;

    const float* x     = (const float*)d_in[0];
    const int*   eidx  = (const int*)d_in[1];
    const float* ew    = (const float*)d_in[2];
    const int*   batch = (const int*)d_in[3];
    const int* src = eidx;
    const int* dst = eidx + EE;

    const float* W[4]  = {(const float*)d_in[4],  (const float*)d_in[8],
                          (const float*)d_in[12], (const float*)d_in[16]};
    const float* gg[4] = {(const float*)d_in[6],  (const float*)d_in[10],
                          (const float*)d_in[14], (const float*)d_in[18]};
    const float* be[4] = {(const float*)d_in[7],  (const float*)d_in[11],
                          (const float*)d_in[15], (const float*)d_in[19]};
    const float* Wf1 = (const float*)d_in[20]; const float* bf1 = (const float*)d_in[21];
    const float* Wf2 = (const float*)d_in[22]; const float* bf2 = (const float*)d_in[23];
    const float* Wf3 = (const float*)d_in[24]; const float* bf3 = (const float*)d_in[25];

    const int Kd[4]  = {1280, 640, 512, 256};
    const int Fod[4] = {640, 512, 256, 256};

    char* ws = (char*)d_ws;
    size_t off = 0;
    __bf16* xbfA = (__bf16*)(ws + off); off += alignUp((size_t)NN * F0 * 2);       // 51.2 MB
    __bf16* xbfB = (__bf16*)(ws + off); off += alignUp((size_t)NN * 640 * 2);      // 25.6 MB
    float*  Hbuf = (float*)(ws + off);  off += alignUp((size_t)NN * 640 * 4);      // 51.2 MB
    float*  Agg  = (float*)(ws + off);  off += alignUp((size_t)NN * 640 * 4);      // 51.2 MB
    __bf16* Wt[4];
    for (int i = 0; i < 4; ++i) {
        Wt[i] = (__bf16*)(ws + off);
        off += alignUp((size_t)Kd[i] * Fod[i] * 2);
    }
    float* sums   = (float*)(ws + off); off += alignUp(640 * 4);
    float* sumsq  = (float*)(ws + off); off += alignUp(640 * 4);
    float* pooled = (float*)(ws + off); off += alignUp((size_t)NG * 256 * 4);

    const int B = 256;

    // prep: bf16 input + transposed bf16 weights
    k_f32_to_bf16<<<gblk((long long)NN * F0, B), B, 0, stream>>>(x, xbfA, (long long)NN * F0);
    for (int i = 0; i < 4; ++i) {
        long long tot = (long long)Kd[i] * Fod[i];
        k_transpose_bf16<<<gblk(tot, B), B, 0, stream>>>(W[i], Wt[i], Kd[i], Fod[i]);
    }

    __bf16* inb  = xbfA;
    __bf16* outb = xbfB;
    for (int L = 0; L < 4; ++L) {
        const int K = Kd[L], Fo = Fod[L];
        const long long nH = (long long)NN * Fo;

        k_zero_f32<<<gblk(nH, B), B, 0, stream>>>(Agg, nH);
        k_zero_f32<<<gblk(Fo, B), B, 0, stream>>>(sums, Fo);
        k_zero_f32<<<gblk(Fo, B), B, 0, stream>>>(sumsq, Fo);

        // 32x64 per wave, 2 waves per block -> block tile 32x128
        dim3 ggrid(NN / 32, Fo / 128);
        k_gemm_bf16<<<ggrid, 64, 0, stream>>>(inb, Wt[L], Hbuf, K, Fo);

        long long sc = (long long)EE * (Fo >> 2);
        k_scatter<<<gblk(sc, B), B, 0, stream>>>(Hbuf, src, dst, ew, Agg, EE, Fo);

        const int chunks = 32;
        const int rowsPer = (NN + chunks - 1) / chunks;
        dim3 cgrid((Fo + B - 1) / B, chunks);
        k_colstats<<<cgrid, B, 0, stream>>>(Agg, sums, sumsq, NN, Fo, rowsPer);

        k_bn_act<<<gblk(nH, B), B, 0, stream>>>(Agg, sums, sumsq, gg[L], be[L],
                                                Hbuf, outb, NN, Fo);
        __bf16* tmp = inb; inb = outb; outb = tmp;
    }

    // pooling over batch ids, then MLP head
    k_zero_f32<<<gblk((long long)NG * 256, B), B, 0, stream>>>(pooled, (long long)NG * 256);
    long long pt = (long long)NN * (256 >> 2);
    k_pool<<<gblk(pt, B), B, 0, stream>>>(Hbuf, batch, pooled, NN, 256);
    k_fc_head<<<NG, 128, 0, stream>>>(pooled, Wf1, bf1, Wf2, bf2, Wf3, bf3, (float*)d_out);
}
